// MannLayer_77919296684675
// MI455X (gfx1250) — compile-verified, matrix operand order checked by reading, not run on hardware
//
#include <hip/hip_runtime.h>
#include <math.h>

// ---------------------------------------------------------------------------
// MANN / NTM layer for gfx1250 (MI455X).  B=16 == WMMA M.  f16 WMMA, f32 acc.
// TDM (tensor_load_to_lds) double-buffers LSTM weight tiles into LDS.
// ---------------------------------------------------------------------------

typedef __attribute__((ext_vector_type(16))) _Float16     v16h;
typedef __attribute__((ext_vector_type(8)))  float        v8f;
typedef __attribute__((ext_vector_type(4)))  unsigned int v4u;
typedef __attribute__((ext_vector_type(8)))  int          v8i;
typedef __attribute__((ext_vector_type(4)))  int          v4i;

#define NB   16      // batch
#define TT   128     // timesteps
#define II   512     // input dim
#define NN   512     // memory slots
#define DD   64      // memory width
#define UU   512     // hidden units
#define PHH  70      // D+6 head params
#define PP   268     // dense projection width
#define KS_A 34      // (576+512)/32 k-steps for LSTM GEMM
#define KS_B 16      // 512/32 k-steps for dense GEMM
#define NT_B 17      // ceil(268/16) N-tiles for dense GEMM

#if __has_builtin(__builtin_amdgcn_tensor_load_to_lds) && \
    __has_builtin(__builtin_amdgcn_s_wait_tensorcnt)    && \
    __has_builtin(__builtin_amdgcn_readfirstlane)
#define HAVE_TDM 1
#else
#define HAVE_TDM 0
#endif

// Per-lane K offsets for the 16-bit 16x32 WMMA operand layout:
// lanes 0-15 hold K {0..7, 16..23}, lanes 16-31 hold K {8..15, 24..31}.
__device__ __forceinline__ int koff_of(int lane, int i) {
    return (lane < 16) ? ((i < 8) ? i : i + 8)
                       : ((i < 8) ? i + 8 : i + 16);
}
__device__ __forceinline__ float sigf(float x)   { return 1.0f / (1.0f + expf(-x)); }
__device__ __forceinline__ float splusf(float x) { return (x > 20.0f) ? x : log1pf(expf(x)); }

// ---------------------------------------------------------------------------
// TDM: async-DMA one contiguous 1KB tile (128 x 8B) from global into LDS.
// Per-wave private: each wave loads only the LDS region it will itself read,
// so the wave's own TENSORcnt is the only synchronization needed.
// ---------------------------------------------------------------------------
__device__ __forceinline__ void tdm_load_1k(const _Float16* gsrc, unsigned int lds_off) {
#if HAVE_TDM
    unsigned long long ga = (unsigned long long)(uintptr_t)gsrc;
    unsigned int g_lo = (unsigned int)ga;
    unsigned int g_hi = (unsigned int)(ga >> 32);
    g_lo    = (unsigned int)__builtin_amdgcn_readfirstlane((int)g_lo);
    g_hi    = (unsigned int)__builtin_amdgcn_readfirstlane((int)g_hi);
    lds_off = (unsigned int)__builtin_amdgcn_readfirstlane((int)lds_off);
    // D# group0: count=1 | lds_addr | global_addr[56:0] | type=2 (bits 127:126)
    v4u g0 = { 1u, lds_off, g_lo, (g_hi & 0x01FFFFFFu) | 0x80000000u };
    // D# group1: data_size=3 (8B); tensor_dim0=128; tile_dim0=128; stride0=128
    v8i g1 = { 0x30000, (int)(128u << 16), 0, (int)(128u << 16), 0, 128, 0, 0 };
    v4i z4 = { 0, 0, 0, 0 };
    asm volatile("" ::: "memory");
#if __clang_major__ >= 23
    v8i z8 = { 0, 0, 0, 0, 0, 0, 0, 0 };
    __builtin_amdgcn_tensor_load_to_lds(g0, g1, z4, z4, z8, 0);
#else
    __builtin_amdgcn_tensor_load_to_lds(g0, g1, z4, z4, 0);
#endif
    asm volatile("" ::: "memory");
#else
    (void)gsrc; (void)lds_off;
#endif
}

// ---------------------------------------------------------------------------
// Init: state buffers + broadcast init_mem.
// ---------------------------------------------------------------------------
__global__ void init_state(const float* __restrict__ init_mem,
                           float* hA, float* hB, float* c,
                           float* rw, float* ww, float* rv, float* mem) {
    int i = blockIdx.x * blockDim.x + threadIdx.x;
    int stride = gridDim.x * blockDim.x;
    for (int e = i; e < NB * NN * DD; e += stride)
        mem[e] = init_mem[e & (NN * DD - 1)];
    if (i < NB * UU) { hA[i] = 0.0f; hB[i] = 0.0f; c[i] = 0.0f; }
    if (i < NB * NN) { rw[i] = 1.0f / NN; ww[i] = 1.0f / NN; }
    if (i < NB * DD) rv[i] = 0.0f;
}

// ---------------------------------------------------------------------------
// Pack [W_lstm ; U_lstm] (K=1088, N=2048, f32) into f16 B-operand tiles,
// keyed (block, kstep, wave) so each wave's per-kstep tile is 1KB contiguous
// (TDM-friendly):  Bp[((bx*34 + kk)*8 + w)*512 + lane*16 + i].
// ---------------------------------------------------------------------------
__global__ void pack_lstm_w(const float* __restrict__ Wl,
                            const float* __restrict__ Ul,
                            _Float16* __restrict__ Bp) {
    int idx = blockIdx.x * blockDim.x + threadIdx.x;          // (bx,kk,w,lane)
    if (idx >= 16 * KS_A * 8 * 32) return;
    int lane = idx & 31;
    int w    = (idx >> 5) & 7;
    int kk   = (idx >> 8) % KS_A;
    int bx   = (idx >> 8) / KS_A;
    int ntile = (w >> 1) * 32 + bx * 2 + (w & 1);             // gate*32 + bx*2 + sub
    int n = ntile * 16 + (lane & 15);
    _Float16* dst = Bp + (size_t)idx * 16;
    for (int i = 0; i < 16; ++i) {
        int K = kk * 32 + koff_of(lane, i);
        float v = (K < 576) ? Wl[(size_t)K * 2048 + n]
                            : Ul[(size_t)(K - 576) * 2048 + n];
        dst[i] = (_Float16)v;
    }
}

// Pack W_dense (K=512, N=268 padded to 272) into f16 B-operand tiles.
__global__ void pack_dense_w(const float* __restrict__ Wd,
                             _Float16* __restrict__ Bp) {
    int idx = blockIdx.x * blockDim.x + threadIdx.x;
    if (idx >= NT_B * KS_B * 32) return;
    int lane = idx & 31;
    int kk   = (idx >> 5) % KS_B;
    int nt   = idx / (KS_B * 32);
    int n    = nt * 16 + (lane & 15);
    _Float16* dst = Bp + (size_t)idx * 16;
    for (int i = 0; i < 16; ++i) {
        int K = kk * 32 + koff_of(lane, i);
        float v = (n < PP) ? Wd[(size_t)K * PP + n] : 0.0f;
        dst[i] = (_Float16)v;
    }
}

// ---------------------------------------------------------------------------
// Kernel A: z = [x_t | rv | h_prev] @ [W_lstm;U_lstm] + b, fused LSTM gates.
// 16 blocks x 256 threads (8 waves).  Block owns 32 hidden units; wave w
// computes gate g=w/2, sub-tile s=w&1.  TDM double-buffers the wave's next
// 1KB weight tile into LDS while the current WMMA chain executes.
// ---------------------------------------------------------------------------
__global__ void lstm_step(const float* __restrict__ x,
                          const _Float16* __restrict__ Bp,
                          const float* __restrict__ b_lstm,
                          const float* __restrict__ hin,
                          float* __restrict__ hout,
                          float* __restrict__ c,
                          const float* __restrict__ rv,
                          int t) {
    __shared__ _Float16 Apack[KS_A * 32 * 16];   // 34 KB: A operand, all k-steps
    __shared__ float    zbuf[4 * 16 * 32];       // 8 KB: z tiles for 4 gates
#if HAVE_TDM
    __shared__ _Float16 Bbuf[2][8 * 512];        // 16 KB: double-buffered B tiles
#endif

    int tid = threadIdx.x;
    // Build A operand (ci = [x_t, rv, h_prev]) in f16, pre-swizzled per-lane.
    for (int e = tid; e < KS_A * 32 * 16; e += 256) {
        int kk = e >> 9, rem = e & 511;
        int lane = rem >> 4, i = rem & 15;
        int m = lane & 15;
        int K = kk * 32 + koff_of(lane, i);
        float v;
        if (K < II)            v = x[((size_t)m * TT + t) * II + K];
        else if (K < II + DD)  v = rv[m * DD + (K - II)];
        else                   v = hin[m * UU + (K - II - DD)];
        Apack[e] = (_Float16)v;
    }
    __syncthreads();

    int wave = tid >> 5, lane = tid & 31;
    int hbase = blockIdx.x * 32;
    int gate = wave >> 1, sub = wave & 1;

    // Wave's weight stream: 34 contiguous 1KB tiles, stride 4096 halves per kk.
    const _Float16* wsrc = Bp + (size_t)blockIdx.x * KS_A * 8 * 512 + (size_t)wave * 512;

#if HAVE_TDM
    unsigned int ldsb0 = (unsigned int)(uintptr_t)&Bbuf[0][wave * 512];
    unsigned int ldsb1 = (unsigned int)(uintptr_t)&Bbuf[1][wave * 512];
    tdm_load_1k(wsrc, ldsb0);                      // prime kk=0 into buffer 0
#endif

    v8f acc = {};
    for (int kk = 0; kk < KS_A; ++kk) {
        v16h b;
#if HAVE_TDM
        if (kk + 1 < KS_A) {
            tdm_load_1k(wsrc + (size_t)(kk + 1) * 4096,
                        ((kk + 1) & 1) ? ldsb1 : ldsb0);
            __builtin_amdgcn_s_wait_tensorcnt(1);  // tile kk has landed
        } else {
            __builtin_amdgcn_s_wait_tensorcnt(0);
        }
        b = *(const v16h*)&Bbuf[kk & 1][wave * 512 + lane * 16];
#else
        b = *(const v16h*)(wsrc + (size_t)kk * 4096 + lane * 16);
#endif
        v16h a = *(const v16h*)&Apack[(kk * 32 + lane) * 16];
        acc = __builtin_amdgcn_wmma_f32_16x16x32_f16(
                  false, a, false, b, (short)0, acc, false, false);
    }

    // C/D layout: lane<16 -> rows 0..7, lane>=16 -> rows 8..15; col = lane%16.
    {
        int col = sub * 16 + (lane & 15);
        int mb  = (lane < 16) ? 0 : 8;
        for (int r = 0; r < 8; ++r)
            zbuf[(gate * 16 + mb + r) * 32 + col] = acc[r];
    }
    __syncthreads();

    // Fused LSTM pointwise: 16 batch x 32 hidden units owned by this block.
    for (int idx = tid; idx < 512; idx += 256) {
        int m = idx >> 5, hl = idx & 31;
        int hu = hbase + hl;
        float zi = zbuf[(0 * 16 + m) * 32 + hl] + b_lstm[        hu];
        float zf = zbuf[(1 * 16 + m) * 32 + hl] + b_lstm[ 512 + hu];
        float zg = zbuf[(2 * 16 + m) * 32 + hl] + b_lstm[1024 + hu];
        float zo = zbuf[(3 * 16 + m) * 32 + hl] + b_lstm[1536 + hu];
        float cn = sigf(zf) * c[m * UU + hu] + sigf(zi) * tanhf(zg);
        float hn = sigf(zo) * tanhf(cn);
        c[m * UU + hu]    = cn;
        hout[m * UU + hu] = hn;
    }
}

// ---------------------------------------------------------------------------
// Kernel B: p = h @ W_dense + b_dense.  1 block x 256 threads (8 waves).
// ---------------------------------------------------------------------------
__global__ void dense_step(const float* __restrict__ h,
                           const _Float16* __restrict__ Bp,
                           const float* __restrict__ b_dense,
                           float* __restrict__ p) {
    __shared__ _Float16 Apack[KS_B * 32 * 16];   // 16 KB
    int tid = threadIdx.x;
    for (int e = tid; e < KS_B * 32 * 16; e += 256) {
        int kk = e >> 9, rem = e & 511;
        int lane = rem >> 4, i = rem & 15;
        int m = lane & 15;
        int K = kk * 32 + koff_of(lane, i);
        Apack[e] = (_Float16)h[m * UU + K];
    }
    __syncthreads();

    int wave = tid >> 5, lane = tid & 31;
    for (int nt = wave; nt < NT_B; nt += 8) {
        v8f acc = {};
        for (int kk = 0; kk < KS_B; ++kk) {
            v16h a = *(const v16h*)&Apack[(kk * 32 + lane) * 16];
            const _Float16* bptr = Bp + ((size_t)(nt * KS_B + kk) * 32 + lane) * 16;
            v16h b = *(const v16h*)bptr;
            if (kk + 1 < KS_B) __builtin_prefetch(bptr + 32 * 16, 0, 0);
            acc = __builtin_amdgcn_wmma_f32_16x16x32_f16(
                      false, a, false, b, (short)0, acc, false, false);
        }
        int col = nt * 16 + (lane & 15);
        int mb  = (lane < 16) ? 0 : 8;
        if (col < PP) {
            float bias = b_dense[col];
            for (int r = 0; r < 8; ++r)
                p[(size_t)(mb + r) * PP + col] = acc[r] + bias;
        }
    }
}

// ---------------------------------------------------------------------------
// NTM addressing (one block == one batch element, 256 threads).
// Cosine sim -> softmax(beta*sim) -> interpolate -> circular shift -> sharpen.
// Result left in lds_w[512] and written back to wprev (ws).
// ---------------------------------------------------------------------------
__device__ void ntm_addressing(const float* __restrict__ ph,   // 70 params
                               float* __restrict__ wprev,      // [512] in ws
                               const float* __restrict__ memb, // [512*64]
                               float* lds_prm, float* lds_a,
                               float* lds_w, float* lds_red) {
    int tid = threadIdx.x;
    for (int i = tid; i < PHH; i += 256) lds_prm[i] = ph[i];
    __syncthreads();

    float kn = 0.0f;
    for (int d = 0; d < DD; ++d) { float kv = lds_prm[d]; kn += kv * kv; }
    kn = fmaxf(sqrtf(kn), 1e-12f);
    float beta  = splusf(lds_prm[64]);
    float gg    = sigf(lds_prm[65]);
    float sm    = fmaxf(lds_prm[66], fmaxf(lds_prm[67], lds_prm[68]));
    float s0 = expf(lds_prm[66] - sm), s1 = expf(lds_prm[67] - sm), s2 = expf(lds_prm[68] - sm);
    float sden = s0 + s1 + s2; s0 /= sden; s1 /= sden; s2 /= sden;
    float gamma = splusf(lds_prm[69]) + 1.0f;

    for (int n = tid; n < NN; n += 256) {
        const float* row = memb + (size_t)n * DD;
        float dot = 0.0f, rn = 0.0f;
        for (int d = 0; d < DD; ++d) { float mv = row[d]; dot += lds_prm[d] * mv; rn += mv * mv; }
        rn = fmaxf(sqrtf(rn), 1e-12f);
        lds_a[n] = beta * (dot / (kn * rn));
    }
    __syncthreads();

    lds_red[tid] = fmaxf(lds_a[tid], lds_a[tid + 256]);
    __syncthreads();
    for (int s = 128; s > 0; s >>= 1) {
        if (tid < s) lds_red[tid] = fmaxf(lds_red[tid], lds_red[tid + s]);
        __syncthreads();
    }
    float mx = lds_red[0];
    __syncthreads();

    float e0 = expf(lds_a[tid] - mx), e1 = expf(lds_a[tid + 256] - mx);
    lds_a[tid] = e0; lds_a[tid + 256] = e1;
    lds_red[tid] = e0 + e1;
    __syncthreads();
    for (int s = 128; s > 0; s >>= 1) {
        if (tid < s) lds_red[tid] += lds_red[tid + s];
        __syncthreads();
    }
    float inv = 1.0f / lds_red[0];
    __syncthreads();

    for (int n = tid; n < NN; n += 256)
        lds_w[n] = gg * (lds_a[n] * inv) + (1.0f - gg) * wprev[n];
    __syncthreads();

    for (int n = tid; n < NN; n += 256) {
        float wsh = s0 * lds_w[(n + NN - 1) & (NN - 1)]
                  + s1 * lds_w[n]
                  + s2 * lds_w[(n + 1) & (NN - 1)];
        lds_a[n] = powf(fmaxf(wsh, 0.0f), gamma);
    }
    __syncthreads();
    lds_red[tid] = lds_a[tid] + lds_a[tid + 256];
    __syncthreads();
    for (int s = 128; s > 0; s >>= 1) {
        if (tid < s) lds_red[tid] += lds_red[tid + s];
        __syncthreads();
    }
    float invs = 1.0f / (lds_red[0] + 1e-8f);
    __syncthreads();
    for (int n = tid; n < NN; n += 256) {
        float wn = lds_a[n] * invs;
        lds_w[n] = wn;
        wprev[n] = wn;
    }
    __syncthreads();
}

// Kernel C: write-head addressing + erase/add memory update (pre-update mem).
__global__ void write_head_step(const float* __restrict__ p,
                                float* __restrict__ ww,
                                float* __restrict__ mem) {
    __shared__ float lds_prm[PHH];
    __shared__ float lds_a[NN];
    __shared__ float lds_w[NN];
    __shared__ float lds_red[256];
    int b = blockIdx.x, tid = threadIdx.x;
    const float* pb = p + (size_t)b * PP;
    float* memb = mem + (size_t)b * NN * DD;

    ntm_addressing(pb + PHH, ww + (size_t)b * NN, memb,
                   lds_prm, lds_a, lds_w, lds_red);

    // erase = sigmoid(p[140:204]), add = tanh(p[204:268]) staged in LDS.
    if (tid < DD) {
        lds_red[tid]      = sigf(pb[2 * PHH + tid]);
        lds_red[64 + tid] = tanhf(pb[2 * PHH + DD + tid]);
    }
    __syncthreads();
    for (int e = tid; e < NN * DD; e += 256) {
        int n = e >> 6, d = e & 63;
        float wn = lds_w[n];
        float m  = memb[e];
        memb[e] = m * (1.0f - wn * lds_red[d]) + wn * lds_red[64 + d];
    }
}

// Kernel D: read-head addressing (post-update mem) + read vector + output.
__global__ void read_head_step(const float* __restrict__ p,
                               float* __restrict__ rw,
                               const float* __restrict__ mem,
                               const float* __restrict__ h,
                               float* __restrict__ rv,
                               float* __restrict__ out,
                               int t) {
    __shared__ float lds_prm[PHH];
    __shared__ float lds_a[NN];
    __shared__ float lds_w[NN];
    __shared__ float lds_red[256];
    int b = blockIdx.x, tid = threadIdx.x;
    const float* memb = mem + (size_t)b * NN * DD;

    ntm_addressing(p + (size_t)b * PP, rw + (size_t)b * NN, memb,
                   lds_prm, lds_a, lds_w, lds_red);

    // rv[d] = sum_n rw[n] * mem[n][d]  (4 chunks x 64 dims -> LDS combine)
    int d = tid & 63, ch = tid >> 6;
    float ps = 0.0f;
    for (int n = ch * 128; n < (ch + 1) * 128; ++n)
        ps += lds_w[n] * memb[(size_t)n * DD + d];
    lds_red[tid] = ps;
    __syncthreads();
    if (tid < DD) {
        float v = lds_red[tid] + lds_red[64 + tid] + lds_red[128 + tid] + lds_red[192 + tid];
        rv[b * DD + tid] = v;
        lds_a[tid] = v;
    }
    __syncthreads();

    float* outp = out + ((size_t)b * TT + t) * (UU + DD);
    for (int j = tid; j < UU + DD; j += 256)
        outp[j] = (j < UU) ? h[(size_t)b * UU + j] : lds_a[j - UU];
}

// ---------------------------------------------------------------------------
// Host launcher.
// ---------------------------------------------------------------------------
extern "C" void kernel_launch(void* const* d_in, const int* in_sizes, int n_in,
                              void* d_out, int out_size, void* d_ws, size_t ws_size,
                              hipStream_t stream) {
    (void)in_sizes; (void)n_in; (void)out_size; (void)ws_size;
    const float* x   = (const float*)d_in[0];
    const float* im  = (const float*)d_in[1];
    const float* Wl  = (const float*)d_in[2];
    const float* Ul  = (const float*)d_in[3];
    const float* bl  = (const float*)d_in[4];
    const float* Wd  = (const float*)d_in[5];
    const float* bd  = (const float*)d_in[6];
    float* out = (float*)d_out;

    char* base = (char*)d_ws;
    size_t off = 0;
    auto take = [&](size_t bytes) -> char* {
        char* pp = base + off;
        off += (bytes + 255) & ~(size_t)255;
        return pp;
    };
    _Float16* BpL = (_Float16*)take((size_t)16 * KS_A * 8 * 32 * 16 * sizeof(_Float16));
    _Float16* BpD = (_Float16*)take((size_t)NT_B * KS_B * 32 * 16 * sizeof(_Float16));
    float* hA  = (float*)take((size_t)NB * UU * 4);
    float* hB  = (float*)take((size_t)NB * UU * 4);
    float* c   = (float*)take((size_t)NB * UU * 4);
    float* mem = (float*)take((size_t)NB * NN * DD * 4);
    float* rw  = (float*)take((size_t)NB * NN * 4);
    float* ww  = (float*)take((size_t)NB * NN * 4);
    float* rv  = (float*)take((size_t)NB * DD * 4);
    float* p   = (float*)take((size_t)NB * PP * 4);

    init_state<<<dim3(1024), dim3(256), 0, stream>>>(im, hA, hB, c, rw, ww, rv, mem);
    pack_lstm_w<<<dim3((16 * KS_A * 8 * 32 + 255) / 256), dim3(256), 0, stream>>>(Wl, Ul, BpL);
    pack_dense_w<<<dim3((NT_B * KS_B * 32 + 255) / 256), dim3(256), 0, stream>>>(Wd, BpD);

    for (int t = 0; t < TT; ++t) {
        const float* hin  = (t & 1) ? hB : hA;
        float*       hout = (t & 1) ? hA : hB;
        lstm_step<<<dim3(16), dim3(256), 0, stream>>>(x, BpL, bl, hin, hout, c, rv, t);
        dense_step<<<dim3(1), dim3(256), 0, stream>>>(hout, BpD, bd, p);
        write_head_step<<<dim3(16), dim3(256), 0, stream>>>(p, ww, mem);
        read_head_step<<<dim3(16), dim3(256), 0, stream>>>(p, rw, mem, hout, rv, out, t);
    }
}
